// HigherOrderSimplicialConv_64123861729553
// MI455X (gfx1250) — matrix operand order repaired and checked by reading, not compiled, not run on hardware
//
#include <hip/hip_runtime.h>
#include <hip/hip_bf16.h>

#define CIN   128
#define COUT  128
#define BN_EPS 1e-5f

typedef __attribute__((ext_vector_type(16))) __bf16 v16bf;
typedef __attribute__((ext_vector_type(8)))  float  v8f;

// ---------------------------------------------------------------------------
// Kernel 0: zero workspace regions (Z_conv accumulators + stat buffers).
// Must run every call: harness does not re-poison between graph replays.
// ---------------------------------------------------------------------------
__global__ void k_zero_f32(float* __restrict__ p, size_t n) {
    size_t i      = (size_t)blockIdx.x * blockDim.x + threadIdx.x;
    size_t stride = (size_t)gridDim.x * blockDim.x;
    for (; i < n; i += stride) p[i] = 0.0f;
}

// ---------------------------------------------------------------------------
// Kernel 1: Z_theta = Z_H @ W + b   via bf16 WMMA with 3-term split precision
// (tf32-class accuracy). One wave per 16x16 output tile, K=128 in 4 steps.
// ---------------------------------------------------------------------------
__global__ void k_gemm_bias_wmma(const float* __restrict__ A,    // [N, CIN]
                                 const float* __restrict__ W,    // [CIN, COUT]
                                 const float* __restrict__ bias, // [COUT]
                                 float* __restrict__ Zt,         // [N, COUT]
                                 int nRows) {
    const int tilesN = COUT / 16;                       // 8
    const int wave   = (blockIdx.x * (blockDim.x >> 5)) + (threadIdx.x >> 5);
    const int lane   = threadIdx.x & 31;
    const int tileM  = wave / tilesN;
    const int tileN  = wave % tilesN;
    if (tileM * 16 >= nRows) return;

    const int mrow  = tileM * 16 + (lane & 15);         // A row this lane holds
    const int khalf = lane >> 4;                        // 0: K low-half, 1: high-half
    const int ncol  = tileN * 16 + (lane & 15);         // B/C column this lane holds
    const int mrowc = (mrow < nRows) ? mrow : 0;

    const float* __restrict__ arow = A + (size_t)mrowc * CIN;

    v8f acc = {};
    #pragma unroll
    for (int k0 = 0; k0 < CIN; k0 += 32) {
        // --- gather fp32 fragments per ISA 16-bit A/B layouts (wave32) ---
        float a_f[16], b_f[16];
        #pragma unroll
        for (int e = 0; e < 8; ++e) a_f[e]     = arow[k0 + khalf * 8 + e];        // K = k0 .. +7 / +8..15
        #pragma unroll
        for (int e = 0; e < 8; ++e) a_f[8 + e] = arow[k0 + 16 + khalf * 8 + e];   // K = k0+16..
        #pragma unroll
        for (int e = 0; e < 16; ++e)
            b_f[e] = W[(size_t)(k0 + khalf * 8 + e + ((e >= 8) ? 8 : 0)) * COUT + ncol]; // B is K x N

        // --- split fp32 = bf16_hi + bf16_lo ---
        v16bf a_hi, a_lo, b_hi, b_lo;
        #pragma unroll
        for (int e = 0; e < 16; ++e) {
            __bf16 ah = (__bf16)a_f[e];
            a_hi[e] = ah;
            a_lo[e] = (__bf16)(a_f[e] - (float)ah);
            __bf16 bh = (__bf16)b_f[e];
            b_hi[e] = bh;
            b_lo[e] = (__bf16)(b_f[e] - (float)bh);
        }

        // acc += A_hi*B_lo + A_lo*B_hi + A_hi*B_hi   (drop lo*lo term)
        acc = __builtin_amdgcn_wmma_f32_16x16x32_bf16(false, a_hi, false, b_lo,
                                                      (short)0, acc, false, false);
        acc = __builtin_amdgcn_wmma_f32_16x16x32_bf16(false, a_lo, false, b_hi,
                                                      (short)0, acc, false, false);
        acc = __builtin_amdgcn_wmma_f32_16x16x32_bf16(false, a_hi, false, b_hi,
                                                      (short)0, acc, false, false);
    }

    // C/D layout: VGPR r, lanes 0-15 -> row r; lanes 16-31 -> row 8+r; col = lane&15
    const float bv = bias[ncol];
    #pragma unroll
    for (int r = 0; r < 8; ++r) {
        int row = tileM * 16 + khalf * 8 + r;
        if (row < nRows) Zt[(size_t)row * COUT + ncol] = acc[r] + bv;
    }
}

// ---------------------------------------------------------------------------
// Kernel 2: COO SpMM scatter: Z_conv[rows[i]] += values[i] * Z_theta[cols[i]]
// One wave per edge; lane handles 4 channels (float4 gather, 4 f32 atomics).
// Z_theta + Z_conv (102 MB total) are L2-resident on the 192 MB L2.
// ---------------------------------------------------------------------------
__global__ void k_spmm_scatter(const float* __restrict__ Zt,   // [N, COUT]
                               const float* __restrict__ vals, // [NNZ]
                               const int*   __restrict__ rows, // [NNZ]
                               const int*   __restrict__ cols, // [NNZ]
                               float* __restrict__ Zc,         // [N, COUT]
                               int nnz) {
    const int wid  = (blockIdx.x * (blockDim.x >> 5)) + (threadIdx.x >> 5);
    const int lane = threadIdx.x & 31;
    if (wid >= nnz) return;

    // Prefetch the gather row of an edge ~64 ahead (gfx1250 global_prefetch_b8).
    const int pf = wid + 64;
    if (pf < nnz) {
        const int cpf = __builtin_nontemporal_load(&cols[pf]);
        __builtin_prefetch(Zt + (size_t)cpf * COUT + lane * 4, 0, 3);
    }

    const int   r = rows[wid];
    const int   c = cols[wid];
    const float v = vals[wid];

    const float4 g = ((const float4*)(Zt + (size_t)c * COUT))[lane];
    float* __restrict__ dst = Zc + (size_t)r * COUT + lane * 4;
    atomicAdd(dst + 0, v * g.x);
    atomicAdd(dst + 1, v * g.y);
    atomicAdd(dst + 2, v * g.z);
    atomicAdd(dst + 3, v * g.w);
}

// ---------------------------------------------------------------------------
// Kernel 3: per-channel sum / sum-of-squares partials -> global atomics.
// Block = 256 threads = 2 sub-rows x 128 channels; ROWS_PER_BLOCK row chunk.
// ---------------------------------------------------------------------------
#define ROWS_PER_BLOCK 256
__global__ void k_bn_stats(const float* __restrict__ Zc,  // [N, COUT]
                           float* __restrict__ stats,     // [2*COUT] sums, sumsq
                           int n) {
    __shared__ float s_sum[256];
    __shared__ float s_sq[256];
    const int tid = threadIdx.x;           // 0..255
    const int ch  = tid & (COUT - 1);
    const int sub = tid >> 7;              // 0 or 1
    const int r0  = blockIdx.x * ROWS_PER_BLOCK;
    const int r1  = (r0 + ROWS_PER_BLOCK < n) ? (r0 + ROWS_PER_BLOCK) : n;

    float sum = 0.0f, sq = 0.0f;
    for (int r = r0 + sub; r < r1; r += 2) {
        float x = Zc[(size_t)r * COUT + ch];
        sum += x;
        sq  += x * x;
    }
    s_sum[tid] = sum;
    s_sq[tid]  = sq;
    __syncthreads();
    if (tid < COUT) {
        atomicAdd(&stats[tid],        s_sum[tid] + s_sum[tid + COUT]);
        atomicAdd(&stats[COUT + tid], s_sq[tid]  + s_sq[tid + COUT]);
    }
}

// ---------------------------------------------------------------------------
// Kernel 4: finalize BN affine: scale = gamma * rsqrt(var+eps), shift = beta - mean*scale
// ---------------------------------------------------------------------------
__global__ void k_bn_final(const float* __restrict__ stats,
                           const float* __restrict__ gamma,
                           const float* __restrict__ beta,
                           float* __restrict__ ss,       // [2*COUT] scale, shift
                           float invN) {
    const int ch = threadIdx.x;
    if (ch >= COUT) return;
    const float mean = stats[ch] * invN;
    const float var  = stats[COUT + ch] * invN - mean * mean;
    const float sc   = gamma[ch] * rsqrtf(var + BN_EPS);
    ss[ch]        = sc;
    ss[COUT + ch] = beta[ch] - mean * sc;
}

// ---------------------------------------------------------------------------
// Kernel 5: out[r] = max_ch relu(Zc[r][ch]*scale[ch]+shift[ch])
// One wave per row; lane holds 4 channels; wave32 shfl_xor reduction.
// ---------------------------------------------------------------------------
__global__ void k_rowmax(const float* __restrict__ Zc,  // [N, COUT]
                         const float* __restrict__ ss,  // [2*COUT]
                         float* __restrict__ out,       // [N]
                         int n) {
    const int wid  = (blockIdx.x * (blockDim.x >> 5)) + (threadIdx.x >> 5);
    const int lane = threadIdx.x & 31;
    if (wid >= n) return;

    const float4 x  = ((const float4*)(Zc + (size_t)wid * COUT))[lane];
    const float4 sc = ((const float4*)ss)[lane];
    const float4 sh = ((const float4*)ss)[32 + lane];

    float m = fmaxf(fmaxf(fmaf(x.x, sc.x, sh.x), fmaf(x.y, sc.y, sh.y)),
                    fmaxf(fmaf(x.z, sc.z, sh.z), fmaf(x.w, sc.w, sh.w)));
    m = fmaxf(m, 0.0f);   // relu folded into the max

    #pragma unroll
    for (int off = 16; off > 0; off >>= 1)
        m = fmaxf(m, __shfl_xor(m, off, 32));

    if (lane == 0) out[wid] = m;
}

// ---------------------------------------------------------------------------
// launcher
// ---------------------------------------------------------------------------
extern "C" void kernel_launch(void* const* d_in, const int* in_sizes, int n_in,
                              void* d_out, int out_size, void* d_ws, size_t ws_size,
                              hipStream_t stream) {
    const float* Z_H    = (const float*)d_in[0];   // [N, CIN]
    const float* values = (const float*)d_in[1];   // [NNZ]
    const float* W      = (const float*)d_in[2];   // [CIN, COUT]
    const float* b      = (const float*)d_in[3];   // [COUT]
    const float* gamma  = (const float*)d_in[4];   // [COUT]
    const float* beta   = (const float*)d_in[5];   // [COUT]
    const int*   rows   = (const int*)d_in[6];     // [NNZ]
    const int*   cols   = (const int*)d_in[7];     // [NNZ]
    float*       outp   = (float*)d_out;           // [N, 1]

    const int N   = in_sizes[0] / CIN;
    const int NNZ = in_sizes[1];

    // Workspace layout (floats): Zt[N*COUT] | Zc[N*COUT] | stats[2*COUT] | ss[2*COUT]
    float* Zt    = (float*)d_ws;
    float* Zc    = Zt + (size_t)N * COUT;
    float* stats = Zc + (size_t)N * COUT;
    float* ss    = stats + 2 * COUT;

    // 0) zero Zc + stats (every call: deterministic across graph replays)
    {
        size_t nz = (size_t)N * COUT + 4 * COUT;
        k_zero_f32<<<2048, 256, 0, stream>>>(Zc, nz);
    }

    // 1) GEMM + bias via WMMA: one wave per 16x16 tile, 8 waves per block
    {
        int tiles  = ((N + 15) / 16) * (COUT / 16);
        int blocks = (tiles + 7) / 8;
        k_gemm_bias_wmma<<<blocks, 256, 0, stream>>>(Z_H, W, b, Zt, N);
    }

    // 2) SpMM scatter: one wave per edge
    {
        int blocks = (NNZ + 7) / 8;
        k_spmm_scatter<<<blocks, 256, 0, stream>>>(Zt, values, rows, cols, Zc, NNZ);
    }

    // 3) BN column stats
    {
        int blocks = (N + ROWS_PER_BLOCK - 1) / ROWS_PER_BLOCK;
        k_bn_stats<<<blocks, 256, 0, stream>>>(Zc, stats, N);
    }

    // 4) finalize scale/shift
    k_bn_final<<<1, COUT, 0, stream>>>(stats, gamma, beta, ss, 1.0f / (float)N);

    // 5) normalize + relu + row-max
    {
        int blocks = (N + 7) / 8;
        k_rowmax<<<blocks, 256, 0, stream>>>(Zc, ss, outp, N);
    }
}
